// GraphAttentionLayer_30193620090900
// MI455X (gfx1250) — matrix operand order repaired and checked by reading, not compiled, not run on hardware
//
#include <hip/hip_runtime.h>
#include <hip/hip_bf16.h>

#define NB   8192
#define FIN  128
#define FOUT 32
#define NROWS (2*8*8192)   /* B*T*N = 131072 */

typedef __attribute__((ext_vector_type(16))) __bf16 v16bf;
typedef __attribute__((ext_vector_type(8)))  float  v8f;

union ABf { v16bf v; unsigned u[8]; uint4 q[2]; };

__device__ __forceinline__ unsigned short f2bf(float f) {
  __bf16 b = (__bf16)f;                      // native v_cvt, RTNE
  return __builtin_bit_cast(unsigned short, b);
}
__device__ __forceinline__ unsigned pk2(float lo, float hi) {
  return (unsigned)f2bf(lo) | ((unsigned)f2bf(hi) << 16);
}

// ---------------------------------------------------------------------------
// K1: h = inp @ W  (bf16 WMMA, f32 accum). Fused epilogue:
//   hbfT : bf16 h stored TRANSPOSED as [bt][o][j]  (j contiguous -> K2's B
//          fragments become two contiguous 16B global loads from L2)
//   s1   : per-node  h[i,:] . a1[i,:]
//   hsum : per-(b,t) sum_j h[j,:]   (global f32 atomics, pre-zeroed)
// 256 threads = 8 waves, each wave owns 16 rows x 32 cols -> block = 128 rows.
// ---------------------------------------------------------------------------
__global__ __launch_bounds__(256) void k_feat(
    const float* __restrict__ inp, const float* __restrict__ W,
    const float* __restrict__ a, unsigned short* __restrict__ hbfT,
    float* __restrict__ s1, float* __restrict__ hsum)
{
  __shared__ unsigned short WbT[FOUT * FIN];  // 8 KB bf16 weights, [o][k]
  __shared__ float s1l[128];
  __shared__ float hsl[FOUT];

  const int t = threadIdx.x;
#pragma unroll
  for (int e = 0; e < 16; ++e) {
    int idx = e * 256 + t;                    // flat over W[k][o]
    int k = idx >> 5, o = idx & 31;
    WbT[o * FIN + k] = f2bf(W[idx]);
  }
  if (t < 128) s1l[t] = 0.f;
  if (t < FOUT) hsl[t] = 0.f;
  __syncthreads();

  const int wv = t >> 5, lane = t & 31;
  const int half = lane >> 4, l16 = lane & 15;
  const long r0 = (long)blockIdx.x * 128;      // global row base of block
  const long tile_r0 = r0 + (long)wv * 16;     // this wave's 16-row tile
  const long rowg = tile_r0 + l16;             // A-frag row for this lane
  const int  bt = (int)(r0 >> 13);

  v8f acc0 = {}; v8f acc1 = {};

#pragma unroll
  for (int ks = 0; ks < 4; ++ks) {
    // ---- A fragment: inp row, K chunks {k0..k0+7, k0+16..k0+23}
    ABf af;
    const int k0 = ks * 32 + half * 8;
    const float4* p0 = (const float4*)(inp + rowg * FIN + k0);
    const float4* p1 = (const float4*)(inp + rowg * FIN + k0 + 16);
    float4 f0 = p0[0], f1 = p0[1], f2 = p1[0], f3 = p1[1];
    af.u[0] = pk2(f0.x, f0.y); af.u[1] = pk2(f0.z, f0.w);
    af.u[2] = pk2(f1.x, f1.y); af.u[3] = pk2(f1.z, f1.w);
    af.u[4] = pk2(f2.x, f2.y); af.u[5] = pk2(f2.z, f2.w);
    af.u[6] = pk2(f3.x, f3.y); af.u[7] = pk2(f3.z, f3.w);

    // ---- B fragments: K chunks {k0..k0+7, k0+16..k0+23} from WbT[o][k] (2x ds_load_b128 each)
#pragma unroll
    for (int ct = 0; ct < 2; ++ct) {
      ABf bfr;
      const int col = ct * 16 + l16;
      const uint4* pb = (const uint4*)&WbT[col * FIN + ks * 32 + half * 8];
      bfr.q[0] = pb[0]; bfr.q[1] = pb[2];
      if (ct == 0)
        acc0 = __builtin_amdgcn_wmma_f32_16x16x32_bf16(
            false, af.v, false, bfr.v, (short)0, acc0, false, false);
      else
        acc1 = __builtin_amdgcn_wmma_f32_16x16x32_bf16(
            false, af.v, false, bfr.v, (short)0, acc1, false, false);
    }
  }

  // ---- epilogue: C layout lane<16: N=lane, M=v ; lane>=16: N=lane-16, M=v+8
  const long ig0 = (tile_r0 & (NB - 1)) + half * 8;  // 8 consecutive nodes
#pragma unroll
  for (int ct = 0; ct < 2; ++ct) {
    const int col = ct * 16 + l16;
    const v8f A = ct ? acc1 : acc0;
    // pack 8 bf16 (consecutive j) and store 16B into hbfT[bt][col][*]
    uint4 pq;
    pq.x = pk2(A[0], A[1]); pq.y = pk2(A[2], A[3]);
    pq.z = pk2(A[4], A[5]); pq.w = pk2(A[6], A[7]);
    *(uint4*)&hbfT[((long)bt * FOUT + col) * NB + ig0] = pq;
#pragma unroll
    for (int v = 0; v < 8; ++v) {
      const int m = v + half * 8;
      const long rg = tile_r0 + m;
      const float val = A[v];
      // a1[i,o] = a[o][i], node index i = rg mod N
      atomicAdd(&s1l[wv * 16 + m], val * a[(long)col * NB + (rg & (NB - 1))]);
      atomicAdd(&hsl[col], val);
    }
  }
  __syncthreads();
  if (t < 128) s1[r0 + t] = s1l[t];
  if (t < FOUT) atomicAdd(&hsum[bt * FOUT + t], hsl[t]);
}

// ---------------------------------------------------------------------------
// K2: s2[bt][i] = a2[i,:] . (mask^T h)[bt][i,:]  -- nbr never materialized.
// Block = 256 thr / 8 waves, output tile 64 rows(i) x 32 cols(o); K-chunk 64
// (2 WMMAs/wave/iter), 128 iterations. Mask tile double-buffered in LDS ->
// ONE barrier per iteration; B fragments read directly from L2 (hbfT).
// grid = (bt=16 fastest, 128 i-tiles) so 16 blocks reuse each adj stripe in L2.
// ---------------------------------------------------------------------------
__global__ __launch_bounds__(256) void k_aggr(
    const int* __restrict__ adj, const unsigned short* __restrict__ hbfT,
    const float* __restrict__ a, float* __restrict__ s2)
{
  __shared__ unsigned short mA[2][64][64];  // bf16 mask tiles [buf][i_local][k]
  __shared__ float s2l[64];

  const int t = threadIdx.x;
  const int bt = blockIdx.x;                   // 0..15
  const long i0 = (long)blockIdx.y * 64;
  if (t < 64) s2l[t] = 0.f;

  const int wv = t >> 5, lane = t & 31;
  const int half = lane >> 4, l16 = lane & 15;
  const int rb = (wv >> 1) * 16;               // row-tile base: 0,16,32,48
  const int cb = (wv & 1) * 16;                // col-tile base: 0,16

  const int lr   = t >> 2;                     // loader: k row 0..63
  const int lc16 = (t & 3) * 16;               // loader: 16 adj columns

  // this lane's h row in hbfT (L2-resident): feature col cb+l16
  const unsigned short* hrow = hbfT + ((long)bt * FOUT + cb + l16) * NB;

  v8f acc = {};

  // ---- preload mask tile j0=0 into buffer 0
  {
    const int* ap = adj + (long)lr * NB + i0 + lc16;
#pragma unroll
    for (int g = 0; g < 4; ++g) {
      int4 v = ((const int4*)ap)[g];
      mA[0][lc16 + 4 * g + 0][lr] = v.x > 0 ? 0x3F80 : 0;
      mA[0][lc16 + 4 * g + 1][lr] = v.y > 0 ? 0x3F80 : 0;
      mA[0][lc16 + 4 * g + 2][lr] = v.z > 0 ? 0x3F80 : 0;
      mA[0][lc16 + 4 * g + 3][lr] = v.w > 0 ? 0x3F80 : 0;
    }
  }
  __syncthreads();

  int buf = 0;
  for (int j0 = 0; j0 < NB; j0 += 64) {
    // ---- stage next mask tile into the other buffer (overlaps with WMMA use)
    if (j0 + 64 < NB) {
      const int nxt = buf ^ 1;
      const int* ap = adj + (long)(j0 + 64 + lr) * NB + i0 + lc16;
      __builtin_prefetch(ap + 64LL * NB, 0, 1);      // global_prefetch_b8, 2 ahead
#pragma unroll
      for (int g = 0; g < 4; ++g) {
        int4 v = ((const int4*)ap)[g];
        mA[nxt][lc16 + 4 * g + 0][lr] = v.x > 0 ? 0x3F80 : 0;
        mA[nxt][lc16 + 4 * g + 1][lr] = v.y > 0 ? 0x3F80 : 0;
        mA[nxt][lc16 + 4 * g + 2][lr] = v.z > 0 ? 0x3F80 : 0;
        mA[nxt][lc16 + 4 * g + 3][lr] = v.w > 0 ? 0x3F80 : 0;
      }
    }

    // ---- consume current buffer: 2 k-steps of 32
#pragma unroll
    for (int kk = 0; kk < 2; ++kk) {
      ABf af, bfr;
      const uint4* pa  = (const uint4*)&mA[buf][rb + l16][kk * 32 + half * 8];
      const uint4* pa2 = (const uint4*)&mA[buf][rb + l16][kk * 32 + half * 8 + 16];
      af.q[0] = pa[0]; af.q[1] = pa2[0];
      const uint4* pb = (const uint4*)(hrow + j0 + kk * 32 + half * 8);
      bfr.q[0] = pb[0]; bfr.q[1] = pb[2];
      acc = __builtin_amdgcn_wmma_f32_16x16x32_bf16(
          false, af.v, false, bfr.v, (short)0, acc, false, false);
    }

    buf ^= 1;
    __syncthreads();   // next buffer staged AND current buffer fully consumed
  }

  // ---- epilogue: reduce acc * a2 into per-row s2 (block owns rows fully)
#pragma unroll
  for (int v = 0; v < 8; ++v) {
    const int m = rb + v + half * 8;
    const int o = cb + l16;
    const long ig = i0 + m;
    const float a2v = a[(long)(FOUT + o) * NB + ig];   // a2[i,o] = a[32+o][i]
    atomicAdd(&s2l[m], acc[v] * a2v);
  }
  __syncthreads();
  if (t < 64) s2[(long)bt * NB + i0 + t] = s2l[t];
}

// ---------------------------------------------------------------------------
// K3: out[bt][i][o] = relu( (s1[bt,i] + s2[bt,i]) * hsum[bt,o] )
// ---------------------------------------------------------------------------
__global__ __launch_bounds__(256) void k_final(
    const float* __restrict__ s1, const float* __restrict__ s2,
    const float* __restrict__ hsum, float* __restrict__ out)
{
  const long idx = (long)blockIdx.x * 256 + threadIdx.x;
  const long row = idx >> 5;                 // 0..131071  == bt*N + i
  const int  o   = (int)(idx & 31);
  const int  bt  = (int)(row >> 13);
  const float s  = s1[row] + s2[row];
  const float v  = s * hsum[bt * FOUT + o];
  out[idx] = v > 0.f ? v : 0.f;
}

extern "C" void kernel_launch(void* const* d_in, const int* in_sizes, int n_in,
                              void* d_out, int out_size, void* d_ws, size_t ws_size,
                              hipStream_t stream) {
  const float* inp = (const float*)d_in[0];   // [2,8,8192,128] f32
  const int*   adj = (const int*)  d_in[1];   // [8192,8192] i32
  const float* W   = (const float*)d_in[2];   // [128,32] f32
  const float* a   = (const float*)d_in[3];   // [64,8192] f32
  float* out = (float*)d_out;                 // [2,8,8192,32] f32

  char* ws = (char*)d_ws;
  unsigned short* hbfT = (unsigned short*)ws;                      // 8 MB bf16 h^T
  float* s1   = (float*)(ws + (size_t)8 * 1024 * 1024);            // 0.5 MB
  float* s2   = (float*)(ws + (size_t)8 * 1024 * 1024 + 512 * 1024);
  float* hsum = (float*)(ws + (size_t)8 * 1024 * 1024 + 1024 * 1024);

  hipMemsetAsync(hsum, 0, 16 * FOUT * sizeof(float), stream);      // capture-safe

  k_feat<<<NROWS / 128, 256, 0, stream>>>(inp, W, a, hbfT, s1, hsum);

  dim3 g2(16, NB / 64);   // bt fastest -> L2 reuse of adj column stripes
  k_aggr<<<g2, 256, 0, stream>>>(adj, hbfT, a, s2);

  k_final<<<NROWS * FOUT / 256, 256, 0, stream>>>(s1, s2, hsum, out);
}
